// Decoder_27470610825494
// MI455X (gfx1250) — matrix operand order repaired and checked by reading, compiled, not run on hardware
//
#include <hip/hip_runtime.h>
#include <hip/hip_bf16.h>
#include <math.h>

typedef __bf16 bf16_t;
typedef __attribute__((ext_vector_type(16))) __bf16 v16bf;
typedef __attribute__((ext_vector_type(8)))  __bf16 v8bf;
typedef __attribute__((ext_vector_type(8)))  float  v8f;

#define S_LEN 64
#define BATCH 64
#define T_LEN 32
#define E_DIM 512
#define H_DIM 1024
#define V_DIM 32000
#define H3    3072
#define KX0   1536   /* E + H */
#define K2H   2048   /* 2*H   */

// ---------------- WMMA fragment loaders (wave32, contiguous k order, same in A and B) ----------------
// A (16x32 bf16, MxK): lanes 0-15 -> M=lane, K=k0..k0+15;
//                      lanes 16-31 -> M=lane-16, K=k0+16..k0+31.
__device__ __forceinline__ v16bf load_a_frag(const bf16_t* base, int row0, int ld, int k0) {
  const int lane = threadIdx.x & 31;
  const int m    = lane & 15;
  const int hi   = lane >> 4;
  const bf16_t* p = base + (size_t)(row0 + m) * ld + k0 + hi * 16;
  union { v16bf v; v8bf h[2]; } u;
  u.h[0] = *(const v8bf*)(p);
  u.h[1] = *(const v8bf*)(p + 8);
  return u.v;
}

// B (32x16 bf16, KxN) stored as weight rows W[g, k] (B[k,n] = W[g0+n, k]):
// lane n = lane&15, lanes 0-15 hold K=k0..k0+15, lanes 16-31 hold K=k0+16..k0+31.
__device__ __forceinline__ v16bf load_b_frag(const bf16_t* w, int g0, int ld, int k0) {
  const int lane = threadIdx.x & 31;
  const int n    = lane & 15;
  const int hi   = lane >> 4;
  return *(const v16bf*)(w + (size_t)(g0 + n) * ld + k0 + hi * 16);
}

__device__ __forceinline__ float sigmf(float x) { return 1.f / (1.f + __expf(-x)); }

// ---------------- fp32 -> bf16 conversion ----------------
__global__ void cvt_bf16(const float* __restrict__ src, bf16_t* __restrict__ dst, long long n) {
  long long i = ((long long)blockIdx.x * blockDim.x + threadIdx.x) * 4;
  if (i + 3 < n) {
    float4 f = *(const float4*)(src + i);
    dst[i + 0] = (bf16_t)f.x;
    dst[i + 1] = (bf16_t)f.y;
    dst[i + 2] = (bf16_t)f.z;
    dst[i + 3] = (bf16_t)f.w;
  } else {
    for (; i < n; ++i) dst[i] = (bf16_t)src[i];
  }
}

// ---------------- hidden init: reshape (4,B,512)->(2,B,1024) is a flat view ----------------
__global__ void init_hidden(const float* __restrict__ eh,
                            float* h0f, float* h1f, bf16_t* h0b, bf16_t* h1b) {
  int i = blockIdx.x * 256 + threadIdx.x;
  if (i < BATCH * H_DIM) {
    float a = eh[i];
    float b = eh[BATCH * H_DIM + i];
    h0f[i] = a; h1f[i] = b;
    h0b[i] = (bf16_t)a; h1b[i] = (bf16_t)b;
  }
}

// ---------------- enc_w[b,s,g] = sum_h enc[s,b,h] * W_att[g,h]  (WMMA GEMM) ----------------
// A rows r = s*B + b over flattened encoder_out (S,B,H); output scattered to (b,s,g).
__global__ void encw_gemm(const bf16_t* __restrict__ encB,
                          const bf16_t* __restrict__ wattB,
                          bf16_t* __restrict__ encw) {
  const int wid  = threadIdx.x >> 5;
  const int lane = threadIdx.x & 31;
  const int r0   = (blockIdx.y * 4 + wid) * 16;
  const int g0   = blockIdx.x * 16;
  v8f acc = {};
  for (int kc = 0; kc < H_DIM; kc += 32) {
    v16bf a = load_a_frag(encB, r0, H_DIM, kc);
    v16bf b = load_b_frag(wattB, g0, H_DIM, kc);
    acc = __builtin_amdgcn_wmma_f32_16x16x32_bf16(false, a, false, b, (short)0, acc, false, false);
  }
  const int n  = lane & 15;
  const int hi = lane >> 4;
  const int g  = g0 + n;
#pragma unroll
  for (int v = 0; v < 8; ++v) {
    int r = r0 + v + hi * 8;           // r = s*B + b
    int s = r >> 6, b = r & 63;
    encw[((size_t)(b * S_LEN + s)) * H_DIM + g] = (bf16_t)acc[v];
  }
}

// ---------------- per-step: embedding gather (tok==nullptr -> start token 0) ----------------
__global__ void gather_emb(const float* __restrict__ table, const int* __restrict__ tok,
                           bf16_t* __restrict__ xb) {
  int b = blockIdx.x;
  int token = tok ? tok[b] : 0;
  const float* row = table + (size_t)token * E_DIM;
  for (int e = threadIdx.x; e < E_DIM; e += blockDim.x)
    xb[(size_t)b * KX0 + e] = (bf16_t)row[e];
}

// ---------------- per-step: scores -> softmax -> context; write masks ----------------
__global__ void step_attn(const bf16_t* __restrict__ encw, const float* __restrict__ h0,
                          const float* __restrict__ enc, bf16_t* __restrict__ xb,
                          bf16_t* __restrict__ yb, float* __restrict__ masks, int t) {
  const int b = blockIdx.x, tid = threadIdx.x;
  __shared__ float sc[S_LEN];
  __shared__ float red[256];
  // scores[s] = enc_w[b,s,:] . h0[b,:]  (4 threads per s)
  const int s = tid >> 2, part = tid & 3;
  const bf16_t* rowp = encw + (size_t)(b * S_LEN + s) * H_DIM;
  const float*  hv   = h0 + (size_t)b * H_DIM;
  float acc = 0.f;
  for (int g = part * 256; g < part * 256 + 256; ++g)
    acc += (float)rowp[g] * hv[g];
  red[tid] = acc;
  __syncthreads();
  if (part == 0) sc[s] = red[tid] + red[tid + 1] + red[tid + 2] + red[tid + 3];
  __syncthreads();
  if (tid < S_LEN) {
    float m = -INFINITY;
    for (int i = 0; i < S_LEN; ++i) m = fmaxf(m, sc[i]);
    red[tid] = __expf(sc[tid] - m);
  }
  __syncthreads();
  if (tid < S_LEN) {
    float sum = 0.f;
    for (int i = 0; i < S_LEN; ++i) sum += red[i];
    float p = red[tid] / sum;
    if (masks) masks[(size_t)tid * (T_LEN * BATCH) + (size_t)t * BATCH + b] = p;
    sc[tid] = p;                       // nobody else touches sc between barriers
  }
  __syncthreads();
  // context[h] = sum_s enc[s,b,h] * p[s] ; write bf16 into x[:,512:] and y[:,1024:]
  for (int h = tid; h < H_DIM; h += 256) {
    float c = 0.f;
    for (int s2 = 0; s2 < S_LEN; ++s2)
      c += enc[(size_t)s2 * (BATCH * H_DIM) + (size_t)b * H_DIM + h] * sc[s2];
    bf16_t cb = (bf16_t)c;
    xb[(size_t)b * KX0 + E_DIM + h]  = cb;
    yb[(size_t)b * K2H + H_DIM + h]  = cb;
  }
}

// ---------------- per-step: one GRU layer (6 WMMA accumulators: r/z/n x gi/gh) ----------------
__global__ void step_gru(const bf16_t* __restrict__ xA, int ldA, int Kx,
                         const bf16_t* __restrict__ hB, const float* __restrict__ hF,
                         const bf16_t* __restrict__ Wih, const bf16_t* __restrict__ Whh,
                         const float* __restrict__ bih, const float* __restrict__ bhh,
                         float* __restrict__ hOutF, bf16_t* __restrict__ hOutB,
                         bf16_t* __restrict__ ybDst) {
  const int wid  = threadIdx.x >> 5;
  const int lane = threadIdx.x & 31;
  const int r0   = wid * 16;               // M-tile: rows of batch
  const int j0   = blockIdx.x * 16;        // output column tile within H
  v8f ir = {}, iz = {}, inn = {}, hr = {}, hz = {}, hn = {};
  for (int kc = 0; kc < Kx; kc += 32) {
    v16bf a  = load_a_frag(xA, r0, ldA, kc);
    v16bf br = load_b_frag(Wih, 0 * H_DIM + j0, Kx, kc);
    v16bf bz = load_b_frag(Wih, 1 * H_DIM + j0, Kx, kc);
    v16bf bn = load_b_frag(Wih, 2 * H_DIM + j0, Kx, kc);
    ir  = __builtin_amdgcn_wmma_f32_16x16x32_bf16(false, a, false, br, (short)0, ir,  false, false);
    iz  = __builtin_amdgcn_wmma_f32_16x16x32_bf16(false, a, false, bz, (short)0, iz,  false, false);
    inn = __builtin_amdgcn_wmma_f32_16x16x32_bf16(false, a, false, bn, (short)0, inn, false, false);
  }
  for (int kc = 0; kc < H_DIM; kc += 32) {
    v16bf a  = load_a_frag(hB, r0, H_DIM, kc);
    v16bf br = load_b_frag(Whh, 0 * H_DIM + j0, H_DIM, kc);
    v16bf bz = load_b_frag(Whh, 1 * H_DIM + j0, H_DIM, kc);
    v16bf bn = load_b_frag(Whh, 2 * H_DIM + j0, H_DIM, kc);
    hr = __builtin_amdgcn_wmma_f32_16x16x32_bf16(false, a, false, br, (short)0, hr, false, false);
    hz = __builtin_amdgcn_wmma_f32_16x16x32_bf16(false, a, false, bz, (short)0, hz, false, false);
    hn = __builtin_amdgcn_wmma_f32_16x16x32_bf16(false, a, false, bn, (short)0, hn, false, false);
  }
  const int n  = lane & 15;
  const int hi = lane >> 4;
  const int j  = j0 + n;
  const float bir = bih[j], biz = bih[H_DIM + j], bin = bih[2 * H_DIM + j];
  const float bhr = bhh[j], bhz = bhh[H_DIM + j], bhn = bhh[2 * H_DIM + j];
#pragma unroll
  for (int v = 0; v < 8; ++v) {
    const int   brow = r0 + v + hi * 8;
    const float hp = hF[(size_t)brow * H_DIM + j];
    const float r  = sigmf(ir[v] + bir + hr[v] + bhr);
    const float z  = sigmf(iz[v] + biz + hz[v] + bhz);
    const float nn = tanhf(inn[v] + bin + r * (hn[v] + bhn));
    const float hN = (1.f - z) * nn + z * hp;
    hOutF[(size_t)brow * H_DIM + j] = hN;
    bf16_t hbb = (bf16_t)hN;
    hOutB[(size_t)brow * H_DIM + j] = hbb;
    if (ybDst) ybDst[(size_t)brow * K2H + j] = hbb;   // logits input cols 0..H-1
  }
}

// ---------------- per-step: logits = [h1,ctx](64x2048) @ Wout^T + bout ----------------
__global__ void step_logits(const bf16_t* __restrict__ yb, const bf16_t* __restrict__ WoutB,
                            const float* __restrict__ bout, float* __restrict__ out) {
  const int wid  = threadIdx.x >> 5;
  const int lane = threadIdx.x & 31;
  const int r0   = wid * 16;
  const int n0   = blockIdx.x * 64;        // 4 N-tiles per block
  v8f acc[4] = {};
  for (int kc = 0; kc < K2H; kc += 32) {
    v16bf a = load_a_frag(yb, r0, K2H, kc);
#pragma unroll
    for (int nt = 0; nt < 4; ++nt) {
      const bf16_t* bp = WoutB + (size_t)(n0 + nt * 16 + (lane & 15)) * K2H
                       + kc + ((lane >> 4) << 4);
      __builtin_prefetch((const void*)(bp + 512), 0, 1);   // stream Wout ahead
      v16bf b = *(const v16bf*)bp;
      acc[nt] = __builtin_amdgcn_wmma_f32_16x16x32_bf16(false, a, false, b, (short)0, acc[nt], false, false);
    }
  }
  const int n  = lane & 15;
  const int hi = lane >> 4;
#pragma unroll
  for (int nt = 0; nt < 4; ++nt) {
    const int g  = n0 + nt * 16 + n;
    const float bb = bout[g];
#pragma unroll
    for (int v = 0; v < 8; ++v) {
      const int brow = r0 + v + hi * 8;
      out[(size_t)brow * V_DIM + g] = acc[nt][v] + bb;
    }
  }
}

// ---------------- warm-step greedy argmax over V ----------------
__global__ void argmax_k(const float* __restrict__ logits, int* __restrict__ tok) {
  const int b = blockIdx.x, tid = threadIdx.x;
  float best = -INFINITY; int bi = 0x7fffffff;
  for (int v = tid; v < V_DIM; v += 256) {
    float x = logits[(size_t)b * V_DIM + v];
    if (x > best || (x == best && v < bi)) { best = x; bi = v; }
  }
  __shared__ float bv[256];
  __shared__ int   bx[256];
  bv[tid] = best; bx[tid] = bi;
  __syncthreads();
  for (int s = 128; s > 0; s >>= 1) {
    if (tid < s) {
      if (bv[tid + s] > bv[tid] || (bv[tid + s] == bv[tid] && bx[tid + s] < bx[tid])) {
        bv[tid] = bv[tid + s]; bx[tid] = bx[tid + s];
      }
    }
    __syncthreads();
  }
  if (tid == 0) tok[b] = bx[0];
}

// ------------------------------- host orchestration -------------------------------
extern "C" void kernel_launch(void* const* d_in, const int* in_sizes, int n_in,
                              void* d_out, int out_size, void* d_ws, size_t ws_size,
                              hipStream_t stream) {
  (void)in_sizes; (void)n_in; (void)out_size; (void)ws_size;
  const float* enc     = (const float*)d_in[0];
  const float* enc_hid = (const float*)d_in[1];
  const float* embed   = (const float*)d_in[2];
  const float* W_att   = (const float*)d_in[3];
  const float* Wih0    = (const float*)d_in[4];
  const float* Whh0    = (const float*)d_in[5];
  const float* bih0    = (const float*)d_in[6];
  const float* bhh0    = (const float*)d_in[7];
  const float* Wih1    = (const float*)d_in[8];
  const float* Whh1    = (const float*)d_in[9];
  const float* bih1    = (const float*)d_in[10];
  const float* bhh1    = (const float*)d_in[11];
  const float* Wout    = (const float*)d_in[12];
  const float* bout    = (const float*)d_in[13];
  const int*   trg     = (const int*)d_in[14];

  char* ws = (char*)d_ws;
  size_t off = 0;
  auto carve = [&](size_t bytes) -> void* {
    void* p = (void*)(ws + off);
    off += (bytes + 255) & ~(size_t)255;
    return p;
  };
  bf16_t* WoutB = (bf16_t*)carve((size_t)V_DIM * K2H * 2);      // 131 MB -> L2-resident
  bf16_t* WattB = (bf16_t*)carve((size_t)H_DIM * H_DIM * 2);
  bf16_t* Wih0B = (bf16_t*)carve((size_t)H3 * KX0 * 2);
  bf16_t* Whh0B = (bf16_t*)carve((size_t)H3 * H_DIM * 2);
  bf16_t* Wih1B = (bf16_t*)carve((size_t)H3 * H_DIM * 2);
  bf16_t* Whh1B = (bf16_t*)carve((size_t)H3 * H_DIM * 2);
  bf16_t* encB  = (bf16_t*)carve((size_t)S_LEN * BATCH * H_DIM * 2);
  bf16_t* encW  = (bf16_t*)carve((size_t)BATCH * S_LEN * H_DIM * 2);
  float*  h0f[2] = { (float*)carve((size_t)BATCH * H_DIM * 4), (float*)carve((size_t)BATCH * H_DIM * 4) };
  float*  h1f[2] = { (float*)carve((size_t)BATCH * H_DIM * 4), (float*)carve((size_t)BATCH * H_DIM * 4) };
  bf16_t* h0b[2] = { (bf16_t*)carve((size_t)BATCH * H_DIM * 2), (bf16_t*)carve((size_t)BATCH * H_DIM * 2) };
  bf16_t* h1b[2] = { (bf16_t*)carve((size_t)BATCH * H_DIM * 2), (bf16_t*)carve((size_t)BATCH * H_DIM * 2) };
  bf16_t* xb  = (bf16_t*)carve((size_t)BATCH * KX0 * 2);
  bf16_t* yb  = (bf16_t*)carve((size_t)BATCH * K2H * 2);
  int*    tok0 = (int*)carve((size_t)BATCH * 4);

  auto cvt = [&](const float* s, bf16_t* d, long long nel) {
    int blocks = (int)((nel / 4 + 255) / 256);
    cvt_bf16<<<blocks, 256, 0, stream>>>(s, d, nel);
  };
  cvt(Wout,  WoutB, (long long)V_DIM * K2H);
  cvt(W_att, WattB, (long long)H_DIM * H_DIM);
  cvt(Wih0,  Wih0B, (long long)H3 * KX0);
  cvt(Whh0,  Whh0B, (long long)H3 * H_DIM);
  cvt(Wih1,  Wih1B, (long long)H3 * H_DIM);
  cvt(Whh1,  Whh1B, (long long)H3 * H_DIM);
  cvt(enc,   encB,  (long long)S_LEN * BATCH * H_DIM);

  init_hidden<<<(BATCH * H_DIM + 255) / 256, 256, 0, stream>>>(enc_hid, h0f[0], h1f[0], h0b[0], h1b[0]);
  encw_gemm<<<dim3(H_DIM / 16, (S_LEN * BATCH) / 64), 128, 0, stream>>>(encB, WattB, encW);

  float* logitsOut = (float*)d_out;                               // (T,B,V)
  float* masksOut  = logitsOut + (size_t)T_LEN * BATCH * V_DIM;   // (S,1,T*B)

  int cur = 0;
  for (int stepi = 0; stepi < T_LEN + 1; ++stepi) {
    const int t   = stepi - 1;            // -1 = warm-up (start token 0)
    const int nxt = cur ^ 1;
    const int* tok = (t < 0)  ? (const int*)nullptr
                   : (t == 0) ? tok0
                              : (trg + (size_t)(t - 1) * BATCH);
    gather_emb<<<BATCH, 256, 0, stream>>>(embed, tok, xb);
    step_attn<<<BATCH, 256, 0, stream>>>(encW, h0f[cur], enc, xb, yb,
                                         (t >= 0) ? masksOut : (float*)nullptr,
                                         (t >= 0) ? t : 0);
    step_gru<<<H_DIM / 16, 128, 0, stream>>>(xb, KX0, KX0, h0b[cur], h0f[cur],
                                             Wih0B, Whh0B, bih0, bhh0,
                                             h0f[nxt], h0b[nxt], (bf16_t*)nullptr);
    step_gru<<<H_DIM / 16, 128, 0, stream>>>(h0b[nxt], H_DIM, H_DIM, h1b[cur], h1f[cur],
                                             Wih1B, Whh1B, bih1, bhh1,
                                             h1f[nxt], h1b[nxt], yb);
    float* lo = (t >= 0) ? (logitsOut + (size_t)t * BATCH * V_DIM)
                         : logitsOut;     // warm logits use t=0 slot, overwritten later
    step_logits<<<V_DIM / 64, 128, 0, stream>>>(yb, WoutB, bout, lo);
    if (t < 0) argmax_k<<<BATCH, 256, 0, stream>>>(logitsOut, tok0);
    cur = nxt;
  }
}
